// PCT_refine_74071005986906
// MI455X (gfx1250) — matrix operand order repaired and checked by reading, not compiled, not run on hardware
//
#include <hip/hip_runtime.h>
#include <math.h>

#define NP 2048
#define NSAMP 16
#define TM 128
#define TN 64
#define TK 32
#define LDSP 40   // padded LDS row stride in halves (80B: 16B-aligned, conflict-reducing)

typedef __attribute__((ext_vector_type(16))) _Float16 v16h;
typedef __attribute__((ext_vector_type(8)))  _Float16 v8h;
typedef __attribute__((ext_vector_type(8)))  float    v8f;
typedef __attribute__((ext_vector_type(4)))  int      int4v;

#if defined(__AMDGCN__) && __has_builtin(__builtin_amdgcn_global_load_async_to_lds_b128) && __has_builtin(__builtin_amdgcn_s_wait_asynccnt)
#define USE_ASYNC_LDS 1
typedef __attribute__((address_space(1))) int4v g_int4;
typedef __attribute__((address_space(3))) int4v l_int4;
#endif

__device__ __forceinline__ float gelu_f(float x) {
  return 0.5f * x * (1.0f + erff(x * 0.7071067811865475f));
}

// ---------------------------------------------------------------------------
// Stride-folding fp32 -> f16 convert/pack:
//   out[r][c] (row-major, Rp x Kp, zero padded) = in[r*rs + c*cs]
// Produces K-contiguous A' (M x K) and B'^T (N x K) operand images so the
// GEMM hot loop is branch-free and all LDS traffic is 16B-granular.
// ---------------------------------------------------------------------------
__global__ void convh_kernel(const float* __restrict__ in, long long inBat,
                             int rs, int cs, _Float16* __restrict__ out,
                             int R, int CK, int Kp, long long outTot) {
  long long t = (long long)blockIdx.x * blockDim.x + threadIdx.x;
  if (t >= outTot) return;
  int r = (int)(t / Kp);
  int c = (int)(t - (long long)r * Kp);
  const float* ib = in + (long long)blockIdx.y * inBat;
  _Float16* ob = out + (long long)blockIdx.y * outTot;
  float v = (r < R && c < CK) ? ib[(long long)r * rs + (long long)c * cs] : 0.f;
  ob[t] = (_Float16)v;
}

// Stage one 128x32(A) + 64x32(B) f16 tile pair into LDS buffer p_.
#if defined(USE_ASYNC_LDS)
#define STAGE_TILE(k0_, p_)                                                    \
  do {                                                                         \
    __builtin_amdgcn_global_load_async_to_lds_b128(                            \
        (g_int4*)(A + (long long)(m0 + ar0) * Kp + ((k0_) + ac0)),             \
        (l_int4*)&As[p_][ar0][ac0], 0, 0);                                     \
    __builtin_amdgcn_global_load_async_to_lds_b128(                            \
        (g_int4*)(A + (long long)(m0 + ar1) * Kp + ((k0_) + ac1)),             \
        (l_int4*)&As[p_][ar1][ac1], 0, 0);                                     \
    __builtin_amdgcn_global_load_async_to_lds_b128(                            \
        (g_int4*)(Bt + (long long)(n0 + br0) * Kp + ((k0_) + bc0)),            \
        (l_int4*)&Bs[p_][br0][bc0], 0, 0);                                     \
  } while (0)
#else
#define STAGE_TILE(k0_, p_)                                                    \
  do {                                                                         \
    *(v8h*)&As[p_][ar0][ac0] =                                                 \
        *(const v8h*)(A + (long long)(m0 + ar0) * Kp + ((k0_) + ac0));         \
    *(v8h*)&As[p_][ar1][ac1] =                                                 \
        *(const v8h*)(A + (long long)(m0 + ar1) * Kp + ((k0_) + ac1));         \
    *(v8h*)&Bs[p_][br0][bc0] =                                                 \
        *(const v8h*)(Bt + (long long)(n0 + br0) * Kp + ((k0_) + bc0));        \
  } while (0)
#endif

// ---------------------------------------------------------------------------
// WMMA GEMM on pre-packed f16 operands: C[m][n] = sum_k A[m][k] * Bt[n][k]
// A: Mp x Kp (K-contiguous), Bt: Np x Kp (K-contiguous). No bounds checks in
// the K loop (operands zero-padded). 256 threads = 8 waves (wave32), wave grid
// 4(M) x 2(N), each wave a 32x32 tile via 2x2 v_wmma_f32_16x16x32_f16.
// Double-buffered global->LDS staging via CDNA5 async b128 copies: tile i+1
// streams in (ASYNCcnt<=3 deferred wait) while tile i is computed.
// ---------------------------------------------------------------------------
__global__ __launch_bounds__(256) void wmma_gemm_kernel(
    const _Float16* __restrict__ A, long long aBat,
    const _Float16* __restrict__ Bt, long long bBat,
    float* __restrict__ C, long long cBat, int ldc, int Kp,
    const float* __restrict__ bias, int biasMode,
    const float* __restrict__ resid, long long rBat, int rld,
    int M, int Nn, int act, int accum) {
  __shared__ _Float16 As[2][TM][LDSP];
  __shared__ _Float16 Bs[2][TN][LDSP];

  const int bz = blockIdx.z;
  A += (long long)bz * aBat;
  Bt += (long long)bz * bBat;
  C += (long long)bz * cBat;
  if (resid) resid += (long long)bz * rBat;

  const int m0 = blockIdx.y * TM;
  const int n0 = blockIdx.x * TN;
  const int tid = threadIdx.x;
  const int lane = tid & 31;
  const int wave = tid >> 5;
  const int wm = wave >> 1;  // 0..3
  const int wn = wave & 1;   // 0..1
  const int ml = lane & 15;
  const int khA = (lane & 16) ? 16 : 0;  // A fragment K-half base
  const int khB = (lane & 16) ? 16 : 0;  // B fragment K-half base

  v8f acc[2][2];
  v8f zero = {};
  acc[0][0] = zero; acc[0][1] = zero; acc[1][0] = zero; acc[1][1] = zero;

  // staging chunk coordinates (8-half = 16B chunks)
  const int ar0 = tid >> 2, ac0 = (tid & 3) << 3;                  // A chunks 0..255
  const int ar1 = (tid + 256) >> 2, ac1 = ((tid + 256) & 3) << 3;  // 256..511
  const int br0 = tid >> 2, bc0 = (tid & 3) << 3;                  // B chunks 0..255

  const int nk = Kp / TK;
  STAGE_TILE(0, 0);  // prologue: tile 0 -> buffer 0
  for (int t = 0; t < nk; ++t) {
    const int p = t & 1;
    const int k0 = t * TK;
    if (t + 1 < nk) {
      STAGE_TILE(k0 + TK, p ^ 1);  // stream next tile into alternate buffer
#if defined(USE_ASYNC_LDS)
      // async loads complete in order: <=3 outstanding => tile t has landed
      __builtin_amdgcn_s_wait_asynccnt(3);
#endif
    } else {
#if defined(USE_ASYNC_LDS)
      __builtin_amdgcn_s_wait_asynccnt(0);
#endif
    }
    __syncthreads();
    if (k0 + 2 * TK < Kp) {
      __builtin_prefetch(A + (long long)(m0 + ar0) * Kp + (k0 + 2 * TK + ac0), 0, 1);
      __builtin_prefetch(Bt + (long long)(n0 + br0) * Kp + (k0 + 2 * TK + bc0), 0, 1);
    }
    // fragments: all contiguous 16B LDS reads (ds_load_b128)
    union { v16h v; v8h h2[2]; } af[2], bf[2];
#pragma unroll
    for (int i = 0; i < 2; ++i) {
      const int row = wm * 32 + i * 16 + ml;
      af[i].h2[0] = *(const v8h*)&As[p][row][khA];
      af[i].h2[1] = *(const v8h*)&As[p][row][khA + 8];
    }
#pragma unroll
    for (int j = 0; j < 2; ++j) {
      const int col = wn * 32 + j * 16 + ml;
      bf[j].h2[0] = *(const v8h*)&Bs[p][col][khB];
      bf[j].h2[1] = *(const v8h*)&Bs[p][col][khB + 8];
    }
#pragma unroll
    for (int i = 0; i < 2; ++i)
#pragma unroll
      for (int j = 0; j < 2; ++j)
        acc[i][j] = __builtin_amdgcn_wmma_f32_16x16x32_f16(
            false, af[i].v, false, bf[j].v, (short)0, acc[i][j], false, false);
    __syncthreads();
  }

#pragma unroll
  for (int i = 0; i < 2; ++i)
#pragma unroll
    for (int j = 0; j < 2; ++j) {
      union { v8f v; float f[8]; } u;
      u.v = acc[i][j];
#pragma unroll
      for (int r = 0; r < 8; ++r) {
        int m = m0 + wm * 32 + i * 16 + r + ((lane >> 4) << 3);
        int n = n0 + wn * 32 + j * 16 + (lane & 15);
        if (m < M && n < Nn) {
          float v = u.f[r];
          long long ci = (long long)m * ldc + n;
          if (accum) v += C[ci];
          if (biasMode == 1) v += bias[m];
          else if (biasMode == 2) v += bias[n];
          if (resid) v += resid[(long long)m * rld + n];
          if (act == 1) v = gelu_f(v);
          C[ci] = v;
        }
      }
    }
}
#undef STAGE_TILE

// ---------------- elementwise / normalization / softmax kernels -------------

__global__ void dw_kernel(const float* __restrict__ x, const float* __restrict__ w,
                          const float* __restrict__ b, float* __restrict__ y,
                          int C, int Npts) {
  long long t = (long long)blockIdx.x * blockDim.x + threadIdx.x;
  long long tot = (long long)C * Npts;
  if (t >= tot) return;
  int c = (int)(t / Npts);
  long long base = (long long)blockIdx.y * tot;
  y[base + t] = x[base + t] * w[c] + b[c];
}

__global__ void ln_col_kernel(float* __restrict__ x, const float* __restrict__ g,
                              const float* __restrict__ b, int C, int Npts) {
  int n = blockIdx.x * blockDim.x + threadIdx.x;
  if (n >= Npts) return;
  float* xb = x + (long long)blockIdx.y * C * Npts;
  float s = 0.f, s2 = 0.f;
  for (int c = 0; c < C; ++c) {
    float v = xb[(long long)c * Npts + n];
    s += v; s2 += v * v;
  }
  float mean = s / (float)C;
  float var = s2 / (float)C - mean * mean;
  float rs = rsqrtf(var + 1e-5f);
  for (int c = 0; c < C; ++c) {
    long long ix = (long long)c * Npts + n;
    xb[ix] = (xb[ix] - mean) * rs * g[c] + b[c];
  }
}

__global__ __launch_bounds__(256) void softmax_row_kernel(float* __restrict__ S,
                                                          int L, float scale) {
  __shared__ float red[256];
  float* p = S + (long long)blockIdx.x * L;
  int tid = threadIdx.x;
  float mx = -3.4e38f;
  for (int j = tid; j < L; j += 256) mx = fmaxf(mx, p[j] * scale);
  red[tid] = mx; __syncthreads();
  for (int s = 128; s > 0; s >>= 1) {
    if (tid < s) red[tid] = fmaxf(red[tid], red[tid + s]);
    __syncthreads();
  }
  mx = red[0]; __syncthreads();
  float sum = 0.f;
  for (int j = tid; j < L; j += 256) {
    float e = expf(p[j] * scale - mx);
    p[j] = e; sum += e;
  }
  red[tid] = sum; __syncthreads();
  for (int s = 128; s > 0; s >>= 1) {
    if (tid < s) red[tid] += red[tid + s];
    __syncthreads();
  }
  float inv = 1.f / red[0];
  for (int j = tid; j < L; j += 256) p[j] *= inv;
}

__global__ void copy_rows_kernel(float* __restrict__ dst, long long dBat, long long dOff,
                                 const float* __restrict__ src, long long sBat,
                                 long long count) {
  long long t = (long long)blockIdx.x * blockDim.x + threadIdx.x;
  if (t >= count) return;
  dst[(long long)blockIdx.y * dBat + dOff + t] = src[(long long)blockIdx.y * sBat + t];
}

__global__ void bcast_rows_kernel(float* __restrict__ dst, long long dBat, long long dOff,
                                  const float* __restrict__ src, long long sBat,
                                  int C, int Npts) {
  long long t = (long long)blockIdx.x * blockDim.x + threadIdx.x;
  long long tot = (long long)C * Npts;
  if (t >= tot) return;
  int c = (int)(t / Npts);
  dst[(long long)blockIdx.y * dBat + dOff + t] = src[(long long)blockIdx.y * sBat + c];
}

// ---------------- LST (local self-attention over kNN boxes) -----------------

__global__ void knn_half_kernel(const float* __restrict__ coarse,
                                float* __restrict__ halfbuf, int Npts) {
  int i = blockIdx.x * blockDim.x + threadIdx.x;
  if (i >= Npts) return;
  const float* xb = coarse + (long long)blockIdx.y * 3 * Npts;
  float xi = xb[i], yi = xb[Npts + i], zi = xb[2 * Npts + i];
  float best[8];
  for (int t = 0; t < 8; ++t) best[t] = 3.4e38f;
  for (int j = 0; j < Npts; ++j) {
    float dx = xb[j] - xi, dy = xb[Npts + j] - yi, dz = xb[2 * Npts + j] - zi;
    float d2 = dx * dx + dy * dy + dz * dz;
    if (d2 < best[7]) {
      int t = 7;
      while (t > 0 && best[t - 1] > d2) { best[t] = best[t - 1]; --t; }
      best[t] = d2;
    }
  }
  halfbuf[(long long)blockIdx.y * Npts + i] = sqrtf(fmaxf(best[7], 0.f));
}

// Replicates stable argsort(in_box * arange)[:, :, :16]:
// zeros first (j==0 or out-of-box, ascending j), then in-box j ascending.
__global__ void select_idx_kernel(const float* __restrict__ coarse,
                                  const float* __restrict__ halfbuf,
                                  int* __restrict__ idx, int Npts) {
  int i = blockIdx.x * blockDim.x + threadIdx.x;
  if (i >= Npts) return;
  const float* xb = coarse + (long long)blockIdx.y * 3 * Npts;
  float hx = halfbuf[(long long)blockIdx.y * Npts + i];
  float xi = xb[i], yi = xb[Npts + i], zi = xb[2 * Npts + i];
  int* out = idx + ((long long)blockIdx.y * Npts + i) * NSAMP;
  int cnt = 0;
  for (int j = 0; j < Npts && cnt < NSAMP; ++j) {
    bool inb = fabsf(xb[j] - xi) <= hx && fabsf(xb[Npts + j] - yi) <= hx &&
               fabsf(xb[2 * Npts + j] - zi) <= hx;
    if (j == 0 || !inb) out[cnt++] = j;
  }
  for (int j = 1; j < Npts && cnt < NSAMP; ++j) {
    bool inb = fabsf(xb[j] - xi) <= hx && fabsf(xb[Npts + j] - yi) <= hx &&
               fabsf(xb[2 * Npts + j] - zi) <= hx;
    if (inb) out[cnt++] = j;
  }
}

__global__ void gather_feat_kernel(const float* __restrict__ f, const int* __restrict__ idx,
                                   float* __restrict__ gf, int C, int Npts) {
  long long t = (long long)blockIdx.x * blockDim.x + threadIdx.x;
  long long M2 = (long long)Npts * NSAMP;
  if (t >= (long long)C * M2) return;
  int c = (int)(t / M2);
  long long r = t - (long long)c * M2;
  gf[t] = f[(long long)c * Npts + idx[r]];
}

__global__ void pos_mlp1_kernel(const float* __restrict__ cb, const int* __restrict__ idx,
                                const float* __restrict__ w, const float* __restrict__ bias,
                                float* __restrict__ h1, int Npts, int D) {
  long long t = (long long)blockIdx.x * blockDim.x + threadIdx.x;
  long long M2 = (long long)Npts * NSAMP;
  if (t >= M2 * D) return;
  long long row = t / D;
  int d = (int)(t - row * D);
  int i = (int)(row >> 4);
  int j = idx[row];
  float gx = cb[j] - cb[i];
  float gy = cb[Npts + j] - cb[Npts + i];
  float gz = cb[2 * Npts + j] - cb[2 * Npts + i];
  float v = gx * w[d * 3 + 0] + gy * w[d * 3 + 1] + gz * w[d * 3 + 2] + bias[d];
  h1[t] = gelu_f(v);
}

__global__ void lst_logits_kernel(const float* __restrict__ h, const float* __restrict__ a2w,
                                  const float* __restrict__ a2b, float* __restrict__ lg,
                                  int D, int M2) {
  int t = blockIdx.x * blockDim.x + threadIdx.x;
  if (t >= M2) return;
  float s = 0.f;
  for (int d = 0; d < D; ++d) s += a2w[d] * h[(long long)d * M2 + t];
  lg[t] = s + a2b[0];
}

__global__ void softmax16_kernel(float* __restrict__ lg, int Npts) {
  int i = blockIdx.x * blockDim.x + threadIdx.x;
  if (i >= Npts) return;
  float* p = lg + (long long)i * NSAMP;
  float mx = -3.4e38f;
  for (int s = 0; s < NSAMP; ++s) mx = fmaxf(mx, p[s]);
  float sum = 0.f;
  for (int s = 0; s < NSAMP; ++s) { float e = expf(p[s] - mx); p[s] = e; sum += e; }
  float inv = 1.f / sum;
  for (int s = 0; s < NSAMP; ++s) p[s] *= inv;
}

__global__ void lst_attn_kernel(const float* __restrict__ gf, const float* __restrict__ attn,
                                float* __restrict__ out, int C, int Npts) {
  long long t = (long long)blockIdx.x * blockDim.x + threadIdx.x;
  if (t >= (long long)C * Npts) return;
  int c = (int)(t / Npts);
  int i = (int)(t - (long long)c * Npts);
  float s = 0.f;
  for (int k = 0; k < NSAMP; ++k)
    s += gf[(long long)c * Npts * NSAMP + (long long)i * NSAMP + k] *
         attn[(long long)i * NSAMP + k];
  out[(long long)c * Npts + i] = s;
}

// ---------------------------------------------------------------------------

extern "C" void kernel_launch(void* const* d_in, const int* in_sizes, int n_in,
                              void* d_out, int out_size, void* d_ws, size_t ws_size,
                              hipStream_t stream) {
  (void)in_sizes; (void)n_in; (void)out_size; (void)ws_size;
  const float* coarse = (const float*)d_in[1];  // (B,3,N)
  const float* feat_g = (const float*)d_in[2];  // (B,512,1)
  int q = 3;
  auto Pn = [&]() { return (const float*)d_in[q++]; };
  struct DWPW { const float *dw_w, *dw_b, *pw_w, *pw_b; };
  auto getDW = [&]() { DWPW p; p.dw_w=Pn(); p.dw_b=Pn(); p.pw_w=Pn(); p.pw_b=Pn(); return p; };
  struct CTP { const float *proj_w,*proj_b,*n13_g,*n13_b,*in_w,*in_b,*out_w,*out_b,
                            *n12_g,*n12_b,*l11_w,*l11_b,*l12_w,*l12_b; };
  auto getCT = [&]() { CTP p; p.proj_w=Pn();p.proj_b=Pn();p.n13_g=Pn();p.n13_b=Pn();
                       p.in_w=Pn();p.in_b=Pn();p.out_w=Pn();p.out_b=Pn();
                       p.n12_g=Pn();p.n12_b=Pn();p.l11_w=Pn();p.l11_b=Pn();
                       p.l12_w=Pn();p.l12_b=Pn(); return p; };
  struct LSTP { const float *p1_w,*p1_b,*p2_w,*p2_b,*a1_w,*a1_b,*a2_w,*a2_b,
                             *f1_w,*f1_b,*f2_w,*f2_b; };
  auto getLS = [&]() { LSTP p; p.p1_w=Pn();p.p1_b=Pn();p.p2_w=Pn();p.p2_b=Pn();
                       p.a1_w=Pn();p.a1_b=Pn();p.a2_w=Pn();p.a2_b=Pn();
                       p.f1_w=Pn();p.f1_b=Pn();p.f2_w=Pn();p.f2_b=Pn(); return p; };

  DWPW conv_x = getDW(), conv_x1 = getDW(), conv_11 = getDW(), conv_1 = getDW();
  CTP sa1 = getCT(), sa2 = getCT(), sa3 = getCT();
  LSTP lst1 = getLS(), lst2 = getLS();
  DWPW conv_ps = getDW(), conv_delta = getDW(), conv_out1 = getDW(), conv_out = getDW();

  // ---- workspace bump allocator (floats) ----
  size_t off = 0;
  auto alloc = [&](size_t n) { float* p = (float*)d_ws + off; off += n; return p; };
  const size_t S512 = (size_t)2 * 512 * NP;
  float* buf_s  = alloc(S512);
  float* buf_o  = alloc(S512);
  float* buf_t  = alloc(S512);
  float* ctA    = alloc(S512);
  float* ctB    = alloc(S512);
  float* lstA   = alloc(S512);
  float* lstB   = alloc(S512);
  float* lstT   = alloc(S512);
  float* tmpA   = alloc(S512);
  float* tmpB   = alloc(S512);
  float* buf_qkv = alloc((size_t)2 * 1536 * NP);
  float* buf_ffh = alloc((size_t)2 * 2048 * NP);
  float* buf_sc  = alloc((size_t)NP * NP);
  float* buf_y   = alloc((size_t)2 * 128 * NP);
  float* buf_y0  = alloc((size_t)2 * 256 * NP);
  float* buf_half = alloc((size_t)2 * NP);
  int*   buf_idx  = (int*)alloc((size_t)2 * NP * NSAMP);
  float* buf_lg   = alloc((size_t)NP * NSAMP);
  float* L1 = alloc((size_t)512 * NP * NSAMP);  // gathered features (per batch)
  float* L2 = alloc((size_t)512 * NP * NSAMP);  // pos-MLP hidden / agg hidden
  float* L3 = alloc((size_t)512 * NP * NSAMP);  // pos (channel-major)
  float* fgA = alloc(1024);
  float* fgB = alloc(1024);
  // f16 operand images: A' up to 2048x512+pad, B'^T up to 32768x512 (+ batches)
  _Float16* hA = (_Float16*)alloc((size_t)1200000);           // 2.4M halves
  _Float16* hB = (_Float16*)alloc((size_t)8500000);           // 17M halves

  float* xout = (float*)d_out;                         // (B,3,N)
  float* y3   = (float*)d_out + (size_t)2 * 3 * NP;    // (B,128,N)

  // GEMM front-end: packs A (strides ars/acs) and B^T (strides bcs/brs) to f16,
  // zero-padded to tile multiples, then launches the branch-free WMMA kernel.
  auto gemm = [&](const float* A, long long aBat, int ars, int acs,
                  const float* Bm, long long bBat, int brs, int bcs,
                  float* C, long long cBat, int ldc,
                  int M, int Nn, int K,
                  const float* bias, int biasMode,
                  const float* resid, long long rBat, int rld,
                  int act, int accum, int batches) {
    int Kp = (K + TK - 1) & ~(TK - 1);
    int Mp = (M + TM - 1) & ~(TM - 1);
    int Np2 = (Nn + TN - 1) & ~(TN - 1);
    int abN = aBat ? batches : 1;
    int bbN = bBat ? batches : 1;
    long long aTot = (long long)Mp * Kp;
    long long bTot = (long long)Np2 * Kp;
    convh_kernel<<<dim3((unsigned)((aTot + 255) / 256), abN), 256, 0, stream>>>(
        A, aBat, ars, acs, hA, M, K, Kp, aTot);
    convh_kernel<<<dim3((unsigned)((bTot + 255) / 256), bbN), 256, 0, stream>>>(
        Bm, bBat, bcs, brs, hB, Nn, K, Kp, bTot);
    dim3 grid(Np2 / TN, Mp / TM, batches);
    wmma_gemm_kernel<<<grid, 256, 0, stream>>>(
        hA, (aBat ? aTot : 0LL), hB, (bBat ? bTot : 0LL),
        C, cBat, ldc, Kp, bias, biasMode, resid, rBat, rld, M, Nn, act, accum);
  };
  auto dw = [&](const float* x, const float* w, const float* b, float* y,
                int C, int Npts, int batches) {
    long long tot = (long long)C * Npts;
    dim3 g((unsigned)((tot + 255) / 256), batches);
    dw_kernel<<<g, 256, 0, stream>>>(x, w, b, y, C, Npts);
  };

  // ---- cross transformer (channel-major (B, dmo, N) throughout) ----
  auto cross = [&](const CTP& p, const float* x, int dm, int dmo, float* out) {
    const int dh = dmo / 4;
    gemm(p.proj_w, 0, dm, 1, x, (long long)dm * NP, NP, 1,
         buf_s, (long long)dmo * NP, NP, dmo, NP, dm, p.proj_b, 1,
         nullptr, 0, 0, 0, 0, 2);
    ln_col_kernel<<<dim3((NP + 255) / 256, 2), 256, 0, stream>>>(buf_s, p.n13_g, p.n13_b, dmo, NP);
    gemm(p.in_w, 0, dmo, 1, buf_s, (long long)dmo * NP, NP, 1,
         buf_qkv, (long long)3 * dmo * NP, NP, 3 * dmo, NP, dmo, p.in_b, 1,
         nullptr, 0, 0, 0, 0, 2);
    float scale = 1.0f / sqrtf((float)dh);
    for (int b = 0; b < 2; ++b)
      for (int h = 0; h < 4; ++h) {
        const float* qp = buf_qkv + (size_t)b * 3 * dmo * NP + (size_t)h * dh * NP;
        const float* kp = qp + (size_t)dmo * NP;
        const float* vp = qp + (size_t)2 * dmo * NP;
        // S = Q^T K  (M=N=2048, K=dh)
        gemm(qp, 0, 1, NP, kp, 0, NP, 1, buf_sc, 0, NP,
             NP, NP, dh, nullptr, 0, nullptr, 0, 0, 0, 0, 1);
        softmax_row_kernel<<<NP, 256, 0, stream>>>(buf_sc, NP, scale);
        // O_h = V S^T  (M=dh, K=2048, N=2048)
        gemm(vp, 0, NP, 1, buf_sc, 0, 1, NP,
             buf_o + (size_t)b * dmo * NP + (size_t)h * dh * NP, 0, NP,
             dh, NP, NP, nullptr, 0, nullptr, 0, 0, 0, 0, 1);
      }
    gemm(p.out_w, 0, dmo, 1, buf_o, (long long)dmo * NP, NP, 1,
         buf_t, (long long)dmo * NP, NP, dmo, NP, dmo, p.out_b, 1,
         buf_s, (long long)dmo * NP, NP, 0, 0, 2);
    ln_col_kernel<<<dim3((NP + 255) / 256, 2), 256, 0, stream>>>(buf_t, p.n12_g, p.n12_b, dmo, NP);
    gemm(p.l11_w, 0, dmo, 1, buf_t, (long long)dmo * NP, NP, 1,
         buf_ffh, 2048LL * NP, NP, 2048, NP, dmo, p.l11_b, 1,
         nullptr, 0, 0, 1, 0, 2);
    gemm(p.l12_w, 0, 2048, 1, buf_ffh, 2048LL * NP, NP, 1,
         out, (long long)dmo * NP, NP, dmo, NP, 2048, p.l12_b, 1,
         buf_t, (long long)dmo * NP, NP, 0, 0, 2);
  };

  // ---- LST block ----
  auto lst = [&](const LSTP& p, const float* feat, float* out) {
    const int M2 = NP * NSAMP;
    dim3 gk((NP + 127) / 128, 2);
    knn_half_kernel<<<gk, 128, 0, stream>>>(coarse, buf_half, NP);
    select_idx_kernel<<<gk, 128, 0, stream>>>(coarse, buf_half, buf_idx, NP);
    for (int b = 0; b < 2; ++b) {
      const float* cb = coarse + (size_t)b * 3 * NP;
      const int* ib = buf_idx + (size_t)b * NP * NSAMP;
      const float* fb = feat + (size_t)b * 512 * NP;
      long long tot = 512LL * M2;
      gather_feat_kernel<<<(unsigned)((tot + 255) / 256), 256, 0, stream>>>(fb, ib, L1, 512, NP);
      pos_mlp1_kernel<<<(unsigned)((tot + 255) / 256), 256, 0, stream>>>(cb, ib, p.p1_w, p.p1_b, L2, NP, 512);
      // pos = p2 @ h1^T  (channel-major 512 x M2)
      gemm(p.p2_w, 0, 512, 1, L2, 0, 1, 512, L3, 0, M2,
           512, M2, 512, p.p2_b, 1, nullptr, 0, 0, 0, 0, 1);
      // h = gelu(a1[:, :512] @ gfeat + a1[:, 512:] @ pos + a1_b)  (split-K)
      gemm(p.a1_w, 0, 1024, 1, L1, 0, M2, 1, L2, 0, M2,
           512, M2, 512, p.a1_b, 1, nullptr, 0, 0, 0, 0, 1);
      gemm(p.a1_w + 512, 0, 1024, 1, L3, 0, M2, 1, L2, 0, M2,
           512, M2, 512, nullptr, 0, nullptr, 0, 0, 1, 1, 1);
      lst_logits_kernel<<<(M2 + 255) / 256, 256, 0, stream>>>(L2, p.a2_w, p.a2_b, buf_lg, 512, M2);
      softmax16_kernel<<<(NP + 255) / 256, 256, 0, stream>>>(buf_lg, NP);
      lst_attn_kernel<<<(unsigned)((512LL * NP + 255) / 256), 256, 0, stream>>>(
          L1, buf_lg, lstT + (size_t)b * 512 * NP, 512, NP);
    }
    gemm(p.f1_w, 0, 512, 1, lstT, 512LL * NP, NP, 1, buf_s, 512LL * NP, NP,
         512, NP, 512, p.f1_b, 1, nullptr, 0, 0, 1, 0, 2);
    gemm(p.f2_w, 0, 512, 1, buf_s, 512LL * NP, NP, 1, out, 512LL * NP, NP,
         512, NP, 512, p.f2_b, 1, lstT, 512LL * NP, NP, 0, 0, 2);
  };

  // ==================== forward graph ====================
  dw(coarse, conv_x.dw_w, conv_x.dw_b, tmpA, 3, NP, 2);
  gemm(conv_x.pw_w, 0, 3, 1, tmpA, 3LL * NP, NP, 1, tmpB, 64LL * NP, NP,
       64, NP, 3, conv_x.pw_b, 1, nullptr, 0, 0, 1, 0, 2);
  dw(tmpB, conv_x1.dw_w, conv_x1.dw_b, tmpA, 64, NP, 2);
  gemm(conv_x1.pw_w, 0, 64, 1, tmpA, 64LL * NP, NP, 1, buf_y, 128LL * NP, NP,
       128, NP, 64, conv_x1.pw_b, 1, nullptr, 0, 0, 0, 0, 2);
  dw(feat_g, conv_11.dw_w, conv_11.dw_b, fgA, 512, 1, 2);
  gemm(conv_11.pw_w, 0, 512, 1, fgA, 512LL, 1, 1, fgB, 256LL, 1,
       256, 1, 512, conv_11.pw_b, 1, nullptr, 0, 0, 1, 0, 2);
  dw(fgB, conv_1.dw_w, conv_1.dw_b, fgA, 256, 1, 2);
  gemm(conv_1.pw_w, 0, 256, 1, fgA, 256LL, 1, 1, fgB, 128LL, 1,
       128, 1, 256, conv_1.pw_b, 1, nullptr, 0, 0, 0, 0, 2);
  {
    long long cnt = 128LL * NP;
    copy_rows_kernel<<<dim3((unsigned)((cnt + 255) / 256), 2), 256, 0, stream>>>(
        buf_y0, 256LL * NP, 0, buf_y, 128LL * NP, cnt);
    bcast_rows_kernel<<<dim3((unsigned)((cnt + 255) / 256), 2), 256, 0, stream>>>(
        buf_y0, 256LL * NP, 128LL * NP, fgB, 128LL, 128, NP);
  }
  cross(sa1, buf_y0, 256, 512, ctA);
  lst(lst1, ctA, lstA);
  cross(sa2, lstA, 512, 512, ctB);
  lst(lst2, ctB, lstB);
  cross(sa3, lstB, 512, 128, ctA);  // ctA reused as (B,128,N)
  dw(ctA, conv_ps.dw_w, conv_ps.dw_b, tmpA, 128, NP, 2);
  gemm(conv_ps.pw_w, 0, 128, 1, tmpA, 128LL * NP, NP, 1, y3, 128LL * NP, NP,
       128, NP, 128, conv_ps.pw_b, 1, nullptr, 0, 0, 0, 0, 2);
  {
    long long cnt = 128LL * NP;
    copy_rows_kernel<<<dim3((unsigned)((cnt + 255) / 256), 2), 256, 0, stream>>>(
        buf_y0, 256LL * NP, 0, y3, 128LL * NP, cnt);
    copy_rows_kernel<<<dim3((unsigned)((cnt + 255) / 256), 2), 256, 0, stream>>>(
        buf_y0, 256LL * NP, 128LL * NP, buf_y, 128LL * NP, cnt);
  }
  dw(buf_y0, conv_delta.dw_w, conv_delta.dw_b, tmpA, 256, NP, 2);
  gemm(conv_delta.pw_w, 0, 256, 1, tmpA, 256LL * NP, NP, 1, tmpB, 128LL * NP, NP,
       128, NP, 256, conv_delta.pw_b, 1, nullptr, 0, 0, 0, 0, 2);
  dw(tmpB, conv_out1.dw_w, conv_out1.dw_b, tmpA, 128, NP, 2);
  gemm(conv_out1.pw_w, 0, 128, 1, tmpA, 128LL * NP, NP, 1, tmpB, 64LL * NP, NP,
       64, NP, 128, conv_out1.pw_b, 1, nullptr, 0, 0, 1, 0, 2);
  dw(tmpB, conv_out.dw_w, conv_out.dw_b, tmpA, 64, NP, 2);
  gemm(conv_out.pw_w, 0, 64, 1, tmpA, 64LL * NP, NP, 1, xout, 3LL * NP, NP,
       3, NP, 64, conv_out.pw_b, 1, coarse, 3LL * NP, NP, 0, 0, 2);
}
